// GPT_74122545594434
// MI455X (gfx1250) — matrix operand order repaired and checked by reading, not compiled, not run on hardware
//
#include <hip/hip_runtime.h>
#include <hip/hip_bf16.h>
#include <stdint.h>

// GPT-2 small config
#define BB   4
#define TT   1024
#define LL   12
#define HH   12
#define DD   768
#define VV   50257
#define HD   64
#define FF   3072
#define MR   (BB * TT)   // 4096 rows

typedef __attribute__((ext_vector_type(16))) __bf16 v16bf;
typedef __attribute__((ext_vector_type(8)))  float  v8f;

__device__ __forceinline__ __bf16 f2bf(float f) {
  unsigned u = __builtin_bit_cast(unsigned, f);
  unsigned r = (u + 0x7FFFu + ((u >> 16) & 1u)) >> 16;
  unsigned short s = (unsigned short)r;
  return __builtin_bit_cast(__bf16, s);
}

__device__ __forceinline__ float gelu_f(float x) {
  float x3 = x * x * x;
  return 0.5f * x * (1.0f + tanhf(0.7978845608028654f * (x + 0.044715f * x3)));
}

// DPP16 cross-lane move (row-local, 16 lanes); CTRL is a DPP control code.
template <int CTRL>
__device__ __forceinline__ float dpp_movf(float x) {
  int i = __builtin_bit_cast(int, x);
  i = __builtin_amdgcn_update_dpp(0, i, CTRL, 0xf, 0xf, true);
  return __builtin_bit_cast(float, i);
}
// 16-lane butterfly reductions entirely in the VALU (no ds_bpermute):
// xor1 = quad_perm[1,0,3,2] (0xB1), xor2 = quad_perm[2,3,0,1] (0x4E),
// then row_half_mirror (0x141) and row_mirror (0x140).
__device__ __forceinline__ float redmax16(float v) {
  v = fmaxf(v, dpp_movf<0xB1>(v));
  v = fmaxf(v, dpp_movf<0x4E>(v));
  v = fmaxf(v, dpp_movf<0x141>(v));
  v = fmaxf(v, dpp_movf<0x140>(v));
  return v;
}
__device__ __forceinline__ float redsum16(float v) {
  v += dpp_movf<0xB1>(v);
  v += dpp_movf<0x4E>(v);
  v += dpp_movf<0x141>(v);
  v += dpp_movf<0x140>(v);
  return v;
}

// ---------------------------------------------------------------------------
// Embedding: h[row] = wte[x[row]] + wpe[row % T]
// ---------------------------------------------------------------------------
__global__ __launch_bounds__(256) void embed_k(const int* __restrict__ x,
                                               const float* __restrict__ wte,
                                               const float* __restrict__ wpe,
                                               float* __restrict__ h) {
  int row = blockIdx.x;
  int tid = threadIdx.x;
  int tok = x[row];
  int t = row % TT;
  const float* we = wte + (size_t)tok * DD;
  const float* wp = wpe + (size_t)t * DD;
  float* hr = h + (size_t)row * DD;
  for (int d = tid; d < DD; d += 256) hr[d] = we[d] + wp[d];
}

// ---------------------------------------------------------------------------
// LayerNorm: one block (256 thr) per row of D=768 (3 elements per thread)
// ---------------------------------------------------------------------------
__global__ __launch_bounds__(256) void ln_k(const float* __restrict__ x,
                                            const float* __restrict__ sc,
                                            const float* __restrict__ bi,
                                            float* __restrict__ out) {
  __shared__ float r1[256];
  __shared__ float r2[256];
  int row = blockIdx.x, tid = threadIdx.x;
  const float* xr = x + (size_t)row * DD;
  float v0 = xr[tid], v1 = xr[tid + 256], v2 = xr[tid + 512];
  r1[tid] = v0 + v1 + v2;
  r2[tid] = v0 * v0 + v1 * v1 + v2 * v2;
  __syncthreads();
  for (int st = 128; st > 0; st >>= 1) {
    if (tid < st) { r1[tid] += r1[tid + st]; r2[tid] += r2[tid + st]; }
    __syncthreads();
  }
  float mu = r1[0] * (1.0f / DD);
  float var = r2[0] * (1.0f / DD) - mu * mu;
  float inv = rsqrtf(var + 1e-5f);
  float* orow = out + (size_t)row * DD;
  orow[tid]       = (v0 - mu) * inv * sc[tid]       + bi[tid];
  orow[tid + 256] = (v1 - mu) * inv * sc[tid + 256] + bi[tid + 256];
  orow[tid + 512] = (v2 - mu) * inv * sc[tid + 512] + bi[tid + 512];
}

// ---------------------------------------------------------------------------
// bf16 WMMA GEMM: C[M,N] = A[M,K] @ B + epilogue
//   BT=false: B is [K,N] row-major.  BT=true: B is [N,K] row-major (B^T).
//   EPI: 0=none, 1=+bias, 2=+bias,GELU, 3=+bias,+resid
//   NCHECK: bounds-check N (lm_head, N=50257)
// Block = 256 thr (8 waves), 128x128 tile; wave = 64x32 (4x2 WMMA accs);
// K-step 32 through LDS (f32 -> bf16 on the fly).
// Software pipeline: next K tile is held in VGPRs while WMMAs run, so global
// load latency hides behind the matrix phase with a single LDS buffer.
// LDS layouts give contiguous fragment reads: As[m][k], Bs[n][k].
// ---------------------------------------------------------------------------
template <int EPI, bool BT, bool NCHECK>
__global__ __launch_bounds__(256) void gemm_k(const float* __restrict__ A,
                                              const float* __restrict__ Bm,
                                              const float* __restrict__ bias,
                                              const float* __restrict__ resid,
                                              float* __restrict__ C,
                                              int M, int N, int K) {
  __shared__ __bf16 As[128][34];
  __shared__ __bf16 Bs[128][36];   // transposed: [n][k]
  const int tid = threadIdx.x;
  const int lane = tid & 31;
  const int w = tid >> 5;
  const int wr = w >> 2, wc = w & 3;
  const int m0 = blockIdx.y * 128;
  const int n0 = blockIdx.x * 128;
  const int ml = lane & 15;
  const int kb_a = (lane < 16) ? 0 : 16;  // A-frag K base (contiguous, matches B)
  const int kb_b = (lane < 16) ? 0 : 16;  // B-frag K base (contiguous, matches A)

  // cooperative-load geometry (constant per thread)
  const int a_row = tid >> 1, a_col = (tid & 1) * 16;        // A: 128x32
  const int bk_row = tid >> 3, bk_col = (tid & 7) * 16;      // B [K,N]: 32x128
  const int bt_n = tid >> 1, bt_kh = (tid & 1) * 16;         // B^T [N,K]
  const float* Ap = A + (size_t)(m0 + a_row) * K + a_col;
  const float* Bp = !BT ? (Bm + (size_t)bk_row * N + n0 + bk_col)
                        : (Bm + (size_t)(n0 + bt_n) * K + bt_kh);
  const bool bvalid = !BT || !NCHECK || (n0 + bt_n) < N;

  v8f acc[4][2];
#pragma unroll
  for (int i = 0; i < 4; i++)
#pragma unroll
    for (int j = 0; j < 2; j++)
#pragma unroll
      for (int r = 0; r < 8; r++) acc[i][j][r] = 0.0f;

  float ar[16], br[16];
  // prologue: fetch K tile 0 into registers
#pragma unroll
  for (int i = 0; i < 16; i++) ar[i] = Ap[i];
  if (!BT) {
#pragma unroll
    for (int i = 0; i < 16; i++) br[i] = Bp[i];
  } else if (bvalid) {
#pragma unroll
    for (int i = 0; i < 16; i++) br[i] = Bp[i];
  } else {
#pragma unroll
    for (int i = 0; i < 16; i++) br[i] = 0.0f;
  }

  for (int k0 = 0; k0 < K; k0 += 32) {
    // stage current tile (registers -> LDS, f32 -> bf16)
#pragma unroll
    for (int i = 0; i < 16; i++) As[a_row][a_col + i] = f2bf(ar[i]);
    if (!BT) {
#pragma unroll
      for (int i = 0; i < 16; i++) Bs[bk_col + i][bk_row] = f2bf(br[i]);
    } else {
#pragma unroll
      for (int i = 0; i < 16; i++) Bs[bt_n][bt_kh + i] = f2bf(br[i]);
    }
    __syncthreads();

    // issue next tile's global loads; they retire during the WMMA phase
    if (k0 + 32 < K) {
#pragma unroll
      for (int i = 0; i < 16; i++) ar[i] = Ap[k0 + 32 + i];
      if (!BT) {
#pragma unroll
        for (int i = 0; i < 16; i++) br[i] = Bp[(size_t)(k0 + 32) * N + i];
      } else if (bvalid) {
#pragma unroll
        for (int i = 0; i < 16; i++) br[i] = Bp[k0 + 32 + i];
      }
      if (k0 + 64 < K) {  // keep L2 warm two tiles ahead
        __builtin_prefetch(Ap + k0 + 64, 0, 1);
        if (!BT) __builtin_prefetch(Bp + (size_t)(k0 + 64) * N, 0, 1);
        else if (bvalid) __builtin_prefetch(Bp + k0 + 64, 0, 1);
      }
    }

    v16bf af[4], bfr[2];
#pragma unroll
    for (int mf = 0; mf < 4; mf++) {
      int m = wr * 64 + mf * 16 + ml;
#pragma unroll
      for (int j = 0; j < 16; j++) {
        int kk = kb_a + j;
        af[mf][j] = As[m][kk];
      }
    }
#pragma unroll
    for (int nf = 0; nf < 2; nf++) {
      int n = wc * 32 + nf * 16 + ml;
#pragma unroll
      for (int j = 0; j < 16; j++) bfr[nf][j] = Bs[n][kb_b + j];
    }
#pragma unroll
    for (int mf = 0; mf < 4; mf++)
#pragma unroll
      for (int nf = 0; nf < 2; nf++)
        acc[mf][nf] = __builtin_amdgcn_wmma_f32_16x16x32_bf16(
            false, af[mf], false, bfr[nf], (short)0, acc[mf][nf], false, false);
    __syncthreads();
  }

  // epilogue (C layout: VGPR r -> rows r / r+8, lane -> column)
#pragma unroll
  for (int mf = 0; mf < 4; mf++) {
#pragma unroll
    for (int nf = 0; nf < 2; nf++) {
      int col = n0 + wc * 32 + nf * 16 + ml;
      if (NCHECK && col >= N) continue;
      float bv = (EPI != 0) ? bias[col] : 0.0f;
#pragma unroll
      for (int r = 0; r < 8; r++) {
        int row = m0 + wr * 64 + mf * 16 + ((lane < 16) ? r : r + 8);
        float v = acc[mf][nf][r] + bv;
        if (EPI == 2) v = gelu_f(v);
        size_t idx = (size_t)row * N + col;
        if (EPI == 3) v += resid[idx];
        C[idx] = v;
      }
    }
  }
}

// ---------------------------------------------------------------------------
// Flash-style causal attention, WMMA for QK^T and PV.
// Grid: (T/128, B*H). Block 256 thr = 8 waves; wave owns 16 query rows.
// qkv layout: [B*T, 3*D]; q at h*64, k at D+h*64, v at 2D+h*64.
// Next K/V tile is register-staged so its global loads overlap the
// softmax + WMMA phase; softmax row reductions run on DPP16 (VALU), not DS.
// ---------------------------------------------------------------------------
__global__ __launch_bounds__(256) void attn_k(const float* __restrict__ qkv,
                                              float* __restrict__ o) {
  __shared__ __bf16 Ks[32][66];
  __shared__ __bf16 Vs[32][66];
  __shared__ __bf16 Ps[8][16][32];

  const int tid = threadIdx.x, lane = tid & 31, w = tid >> 5;
  const int ml = lane & 15;
  const int qb = blockIdx.x;
  const int bh = blockIdx.y;
  const int b = bh / HH, h = bh % HH;
  const int q0 = qb * 128;
  const int qrow0 = q0 + w * 16;
  const int kb_a = (lane < 16) ? 0 : 16;
  const int kb_b = (lane < 16) ? 0 : 16;

  const size_t rs = 3 * DD;
  const float* qp = qkv + (size_t)b * TT * rs + h * HD;
  const float* kp = qp + DD;
  const float* vp = qp + 2 * DD;

  // Q fragments for this wave's 16 rows (K dim 64 -> 2 frags)
  v16bf aq[2];
  {
    const float* qrow = qp + (size_t)(qrow0 + ml) * rs;
#pragma unroll
    for (int f = 0; f < 2; f++)
#pragma unroll
      for (int j = 0; j < 16; j++) {
        int kk = 32 * f + kb_a + j;
        aq[f][j] = f2bf(qrow[kk]);
      }
  }

  v8f oacc[4];
#pragma unroll
  for (int nf = 0; nf < 4; nf++)
#pragma unroll
    for (int r = 0; r < 8; r++) oacc[nf][r] = 0.0f;
  float mrow[8], lrow[8];
#pragma unroll
  for (int r = 0; r < 8; r++) { mrow[r] = -3e38f; lrow[r] = 0.0f; }

  const int ntiles = (qb + 1) * 4;  // keys 0 .. q0+127 in tiles of 32
  const float scale = 0.125f;       // 1/sqrt(64)

  // cooperative K/V tile geometry: 32x64 f32, 8 per thread
  const int t_row = tid >> 3, t_col = (tid & 7) * 8;
  const float* kbase = kp + (size_t)t_row * rs + t_col;
  const float* vbase = vp + (size_t)t_row * rs + t_col;

  float kr[8], vr[8];
#pragma unroll
  for (int i = 0; i < 8; i++) kr[i] = kbase[i];
#pragma unroll
  for (int i = 0; i < 8; i++) vr[i] = vbase[i];

  for (int ct = 0; ct < ntiles; ct++) {
    int c0 = ct * 32;
    // stage current K/V tile (registers -> LDS, f32 -> bf16)
#pragma unroll
    for (int i = 0; i < 8; i++) Ks[t_row][t_col + i] = f2bf(kr[i]);
#pragma unroll
    for (int i = 0; i < 8; i++) Vs[t_row][t_col + i] = f2bf(vr[i]);
    __syncthreads();

    // issue next tile's global loads; they retire during compute
    if (ct + 1 < ntiles) {
      size_t adv = (size_t)(c0 + 32) * rs;
#pragma unroll
      for (int i = 0; i < 8; i++) kr[i] = kbase[adv + i];
#pragma unroll
      for (int i = 0; i < 8; i++) vr[i] = vbase[adv + i];
    }

    if (c0 <= qrow0 + 15) {  // wave-uniform causal skip
      // S = Q @ K^T : two 16(keys) tiles, K dim 64 -> 2 wmma each
      v8f s[2];
#pragma unroll
      for (int st = 0; st < 2; st++) {
#pragma unroll
        for (int r = 0; r < 8; r++) s[st][r] = 0.0f;
#pragma unroll
        for (int f = 0; f < 2; f++) {
          v16bf bk;
#pragma unroll
          for (int j = 0; j < 16; j++)
            bk[j] = Ks[st * 16 + ml][32 * f + kb_b + j];
          s[st] = __builtin_amdgcn_wmma_f32_16x16x32_bf16(
              false, aq[f], false, bk, (short)0, s[st], false, false);
        }
      }
      // online softmax per output row (C-layout: VGPR r -> row r / r+8)
#pragma unroll
      for (int r = 0; r < 8; r++) {
        int row = qrow0 + ((lane < 16) ? r : r + 8);
        float s0 = s[0][r] * scale, s1 = s[1][r] * scale;
        if (c0 + ml > row)      s0 = -3e38f;
        if (c0 + 16 + ml > row) s1 = -3e38f;
        float mx = redmax16(fmaxf(s0, s1));
        float mnew = fmaxf(mrow[r], mx);
        float alpha = __expf(mrow[r] - mnew);
        float p0 = __expf(s0 - mnew), p1 = __expf(s1 - mnew);
        float ps = redsum16(p0 + p1);
        lrow[r] = lrow[r] * alpha + ps;
        mrow[r] = mnew;
#pragma unroll
        for (int nf = 0; nf < 4; nf++) oacc[nf][r] *= alpha;
        int prow = (lane < 16) ? r : r + 8;
        Ps[w][prow][ml]      = f2bf(p0);
        Ps[w][prow][16 + ml] = f2bf(p1);
      }
      // O += P @ V  (same-wave DS write->read is in order)
      v16bf ap;
#pragma unroll
      for (int j = 0; j < 16; j++) {
        int kk = kb_a + j;
        ap[j] = Ps[w][ml][kk];
      }
#pragma unroll
      for (int nf = 0; nf < 4; nf++) {
        v16bf bv;
#pragma unroll
        for (int j = 0; j < 16; j++) bv[j] = Vs[kb_b + j][nf * 16 + ml];
        oacc[nf] = __builtin_amdgcn_wmma_f32_16x16x32_bf16(
            false, ap, false, bv, (short)0, oacc[nf], false, false);
      }
    }
    __syncthreads();
  }

#pragma unroll
  for (int nf = 0; nf < 4; nf++) {
    int col = h * HD + nf * 16 + ml;
#pragma unroll
    for (int r = 0; r < 8; r++) {
      int row = qrow0 + ((lane < 16) ? r : r + 8);
      o[(size_t)(b * TT + row) * DD + col] = oacc[nf][r] / lrow[r];
    }
  }
}

// ---------------------------------------------------------------------------
// Per-row cross-entropy: rl[row] = logsumexp(logits[row]) - logits[row][y]
// ---------------------------------------------------------------------------
__global__ __launch_bounds__(256) void rowloss_k(const float* __restrict__ logits,
                                                 const int* __restrict__ y,
                                                 float* __restrict__ rl) {
  __shared__ float red[256];
  int row = blockIdx.x, tid = threadIdx.x;
  const float* lr = logits + (size_t)row * VV;
  float mx = -3e38f;
  for (int i = tid; i < VV; i += 256) mx = fmaxf(mx, lr[i]);
  red[tid] = mx;
  __syncthreads();
  for (int st = 128; st > 0; st >>= 1) {
    if (tid < st) red[tid] = fmaxf(red[tid], red[tid + st]);
    __syncthreads();
  }
  mx = red[0];
  __syncthreads();
  float s = 0.0f;
  for (int i = tid; i < VV; i += 256) s += __expf(lr[i] - mx);
  red[tid] = s;
  __syncthreads();
  for (int st = 128; st > 0; st >>= 1) {
    if (tid < st) red[tid] += red[tid + st];
    __syncthreads();
  }
  if (tid == 0) {
    float lse = mx + logf(red[0]);
    rl[row] = lse - lr[y[row]];
  }
}

// deterministic fixed-order final reduction
__global__ __launch_bounds__(256) void lossred_k(const float* __restrict__ rl,
                                                 float* __restrict__ out) {
  __shared__ float red[256];
  int tid = threadIdx.x;
  float s = 0.0f;
  for (int i = tid; i < MR; i += 256) s += rl[i];
  red[tid] = s;
  __syncthreads();
  for (int st = 128; st > 0; st >>= 1) {
    if (tid < st) red[tid] += red[tid + st];
    __syncthreads();
  }
  if (tid == 0) out[0] = red[0] / (float)MR;
}

// ---------------------------------------------------------------------------
extern "C" void kernel_launch(void* const* d_in, const int* in_sizes, int n_in,
                              void* d_out, int out_size, void* d_ws, size_t ws_size,
                              hipStream_t stream) {
  (void)in_sizes; (void)n_in; (void)out_size; (void)ws_size;
  const int*   x      = (const int*)  d_in[0];
  const int*   y      = (const int*)  d_in[1];
  const float* wte    = (const float*)d_in[2];
  const float* wpe    = (const float*)d_in[3];
  const float* ln1_s  = (const float*)d_in[4];
  const float* ln1_b  = (const float*)d_in[5];
  const float* attn_w = (const float*)d_in[6];
  const float* attn_b = (const float*)d_in[7];
  const float* proj_w = (const float*)d_in[8];
  const float* proj_b = (const float*)d_in[9];
  const float* ln2_s  = (const float*)d_in[10];
  const float* ln2_b  = (const float*)d_in[11];
  const float* fc_w   = (const float*)d_in[12];
  const float* fc_b   = (const float*)d_in[13];
  const float* fc2_w  = (const float*)d_in[14];
  const float* fc2_b  = (const float*)d_in[15];
  const float* lnf_s  = (const float*)d_in[16];
  const float* lnf_b  = (const float*)d_in[17];

  float* ws = (float*)d_ws;
  size_t off = 0;
  float* h    = ws + off; off += (size_t)MR * DD;
  float* a    = ws + off; off += (size_t)MR * DD;
  float* qkv  = ws + off; off += (size_t)MR * 3 * DD;
  float* ob   = ws + off; off += (size_t)MR * DD;
  float* mbuf = ws + off; off += (size_t)MR * FF;
  float* rl   = ws + off; off += (size_t)MR;

  float* logits = (float*)d_out;
  float* loss   = logits + (size_t)MR * VV;

  embed_k<<<MR, 256, 0, stream>>>(x, wte, wpe, h);

  for (int l = 0; l < LL; l++) {
    ln_k<<<MR, 256, 0, stream>>>(h, ln1_s + l * DD, ln1_b + l * DD, a);
    gemm_k<1, false, false><<<dim3(3 * DD / 128, MR / 128), 256, 0, stream>>>(
        a, attn_w + (size_t)l * DD * 3 * DD, attn_b + (size_t)l * 3 * DD,
        nullptr, qkv, MR, 3 * DD, DD);
    attn_k<<<dim3(TT / 128, BB * HH), 256, 0, stream>>>(qkv, ob);
    gemm_k<3, false, false><<<dim3(DD / 128, MR / 128), 256, 0, stream>>>(
        ob, proj_w + (size_t)l * DD * DD, proj_b + (size_t)l * DD,
        h, h, MR, DD, DD);
    ln_k<<<MR, 256, 0, stream>>>(h, ln2_s + l * DD, ln2_b + l * DD, a);
    gemm_k<2, false, false><<<dim3(FF / 128, MR / 128), 256, 0, stream>>>(
        a, fc_w + (size_t)l * DD * FF, fc_b + (size_t)l * FF,
        nullptr, mbuf, MR, FF, DD);
    gemm_k<3, false, false><<<dim3(DD / 128, MR / 128), 256, 0, stream>>>(
        mbuf, fc2_w + (size_t)l * FF * DD, fc2_b + (size_t)l * DD,
        h, h, MR, DD, FF);
  }

  ln_k<<<MR, 256, 0, stream>>>(h, lnf_s, lnf_b, a);
  gemm_k<0, true, true><<<dim3((VV + 127) / 128, MR / 128), 256, 0, stream>>>(
      a, wte, nullptr, nullptr, logits, MR, VV, DD);
  rowloss_k<<<MR, 256, 0, stream>>>(logits, y, rl);
  lossred_k<<<1, 256, 0, stream>>>(rl, loss);
}
